// HTSubTree_2413771621186
// MI455X (gfx1250) — matrix operand order repaired and checked by reading, not compiled, not run on hardware
//
#include <hip/hip_runtime.h>

// CDNA5 (gfx1250) wave32 WMMA types
typedef __attribute__((ext_vector_type(16))) __bf16 v16bf;
typedef __attribute__((ext_vector_type(2)))  __bf16 v2bf;
typedef __attribute__((ext_vector_type(8)))  float  v8f;
typedef __attribute__((ext_vector_type(2)))  float  v2f;

union FragAB { v16bf v; uint4 q[2]; };
union AccU   { v8f v; float4 f4[2]; };

// scalar f32 -> bf16 via native fptrunc (lets backend pick v_cvt_pk_bf16_f32)
__device__ inline unsigned short f2bf(float f) {
  __bf16 h = (__bf16)f;
  return __builtin_bit_cast(unsigned short, h);
}

// pack two f32 -> two bf16 in one dword
__device__ inline unsigned int pk_bf16(float a, float b) {
#if __has_builtin(__builtin_amdgcn_cvt_pk_bf16_f32)
  auto pk = __builtin_amdgcn_cvt_pk_bf16_f32(a, b);
  return __builtin_bit_cast(unsigned int, pk);
#else
  v2f s; s[0] = a; s[1] = b;
  v2bf pk = __builtin_convertvector(s, v2bf);
  return __builtin_bit_cast(unsigned int, pk);
#endif
}

// pack v8f accumulator -> 8 bf16 (16 bytes)
__device__ inline uint4 pk8_bf16(v8f a) {
  uint4 r;
  r.x = pk_bf16(a[0], a[1]);
  r.y = pk_bf16(a[2], a[3]);
  r.z = pk_bf16(a[4], a[5]);
  r.w = pk_bf16(a[6], a[7]);
  return r;
}

__device__ inline v8f vzero() {
  v8f z; for (int i = 0; i < 8; ++i) z[i] = 0.f; return z;
}

// A-fragment (16x32 bf16), source row-major [m][k], row stride S elements.
// ISA 16-bit A 16x32 layout: lane m=lo; V0..3 = K(hi*8..+7), V4..7 = K(16+hi*8..+7).
__device__ inline v16bf load_frag_a(const unsigned short* src, int S,
                                    int m0, int k0, int lo, int hi) {
  FragAB f;
  const unsigned short* p = src + (m0 + lo) * S + k0 + hi * 8;
  f.q[0] = *(const uint4*)(p);
  f.q[1] = *(const uint4*)(p + 16);
  return f.v;
}

// B-fragment (32x16 bf16), source stored TRANSPOSED [n][k], row stride S.
// Layout: lane n=lo; V0..3 = K(hi*8..+7), V4..7 = K(16+hi*8..+7).
__device__ inline v16bf load_frag_b(const unsigned short* srcT, int S,
                                    int k0, int n0, int lo, int hi) {
  FragAB f;
  const unsigned short* p = srcT + (n0 + lo) * S + k0 + hi * 8;
  f.q[0] = *(const uint4*)(p);
  f.q[1] = *(const uint4*)(p + 16);
  return f.v;
}

// ---------------------------------------------------------------------------
// Fold factors+cores into the two stationary operator matrices (bf16):
//   P_T[m=(o0,c,o1)][k=(i0,i1)]                  (512x64)
//   W2T[n=(o2,o3,r)][k2=(rep,c,q_local)]         (512x512), n-major
// where q=(i2,i3), rep=q>>5, q_local=q&31  (matches stage-2 reduction order)
// ---------------------------------------------------------------------------
__global__ __launch_bounds__(512) void ht_precompute(
    const float* __restrict__ factors,   // [4][8][8][8] = (p, in, out, rank)
    const float* __restrict__ cores,     // [3][8][8][8]
    unsigned short* __restrict__ P_T,
    unsigned short* __restrict__ W2T) {
  __shared__ float B1s[4096];  // [a][i1][o1][c]
  __shared__ float A1s[4096];  // [e][d][i3][o3]
  const int tid = threadIdx.x;
  const float* f0 = factors;
  const float* f1 = factors + 512;
  const float* f2 = factors + 1024;
  const float* f3 = factors + 1536;
  const float* core0 = cores;          // (0,4): [c][d][r]
  const float* core1 = cores + 512;    // (0,2): [a][bp][c]
  const float* core2 = cores + 1024;   // (2,4): [e][g][d]

  for (int t = tid; t < 4096; t += 512) {
    int a = t >> 9, i1 = (t >> 6) & 7, o1 = (t >> 3) & 7, c = t & 7;
    float s = 0.f;
    for (int bp = 0; bp < 8; ++bp)
      s += core1[a * 64 + bp * 8 + c] * f1[i1 * 64 + o1 * 8 + bp];
    B1s[t] = s;
  }
  for (int t = tid; t < 4096; t += 512) {
    int e = t >> 9, d = (t >> 6) & 7, i3 = (t >> 3) & 7, o3 = t & 7;
    float s = 0.f;
    for (int g = 0; g < 8; ++g)
      s += core2[e * 64 + g * 8 + d] * f3[i3 * 64 + o3 * 8 + g];
    A1s[t] = s;
  }
  __syncthreads();

  for (int t = tid; t < 32768; t += 512) {           // P_T[m][k]
    int m = t >> 6, k = t & 63;
    int o0 = m >> 6, c = (m >> 3) & 7, o1 = m & 7;
    int i0 = k >> 3, i1 = k & 7;
    float s = 0.f;
    for (int a = 0; a < 8; ++a)
      s += f0[i0 * 64 + o0 * 8 + a] * B1s[a * 512 + i1 * 64 + o1 * 8 + c];
    P_T[t] = f2bf(s);
  }
  for (int t = tid; t < 262144; t += 512) {          // W2T[n][k2]
    int n = t >> 9, k2 = t & 511;
    int o2 = n >> 6, o3 = (n >> 3) & 7, r = n & 7;
    int rep = k2 >> 8, c = (k2 >> 5) & 7, ql = k2 & 31;
    int q = rep * 32 + ql;
    int i2 = q >> 3, i3 = q & 7;
    float s = 0.f;
    for (int d = 0; d < 8; ++d) {
      float cr = core0[c * 64 + d * 8 + r];
      for (int e = 0; e < 8; ++e)
        s += cr * f2[i2 * 64 + o2 * 8 + e] * A1s[e * 512 + d * 64 + i3 * 8 + o3];
    }
    W2T[t] = f2bf(s);
  }
}

// ---------------------------------------------------------------------------
// Fused per-batch forward: one workgroup (16 waves, wave32) per batch element.
//   Stage 1: C1[q, m] = XT(64x64) x P (K=64)       -> LDS T3A[(o0,o1)][c,ql] bf16
//   Stage 2: OUT^T[n, m2] = W2T(512xK) x T3A (K=512), f32 direct to global
// Split into two q-halves so LDS = 8KB (XT) + 32KB (T3A half) = 40KB.
// ---------------------------------------------------------------------------
__global__ __launch_bounds__(512) void ht_forward(
    const float* __restrict__ x,
    const unsigned short* __restrict__ P_T,   // [512][64]
    const unsigned short* __restrict__ W2T,   // [512][512]
    float* __restrict__ out) {
  __shared__ unsigned short XT[64 * 64];     // [q=(i2,i3)][k=(i0,i1)]
  __shared__ unsigned short T3A[64 * 256];   // [(o0,o1)][c*32 + q_local]

  const int b    = blockIdx.x;
  const int tid  = threadIdx.x;
  const int wave = tid >> 5;        // 0..15
  const int lane = tid & 31;
  const int lo = lane & 15, hi = lane >> 4;

  // x_b fp32 -> bf16, transposed into LDS
  const float* xb = x + (size_t)b * 4096;
  for (int t = tid; t < 4096; t += 512) {
    int k = t >> 6, q = t & 63;
    XT[q * 64 + k] = f2bf(xb[t]);
  }
  __syncthreads();

  v8f acc2[2][4];   // [n-tile][m2-tile]
  for (int i = 0; i < 2; ++i) for (int j = 0; j < 4; ++j) acc2[i][j] = vzero();

  for (int rep = 0; rep < 2; ++rep) {
    // ---- Stage 1 partial: q in [rep*32, rep*32+32); wave owns m-cols [wave*32,+32)
    v16bf xfr[2][2];                       // [kk][qt]
    for (int kk2 = 0; kk2 < 2; ++kk2)
      for (int qt = 0; qt < 2; ++qt)
        xfr[kk2][qt] = load_frag_a(XT, 64, rep * 32 + qt * 16, kk2 * 32, lo, hi);

    for (int mi = 0; mi < 2; ++mi) {
      int mbase = (wave * 2 + mi) * 16;
      v8f a1[2];
      a1[0] = vzero(); a1[1] = vzero();
      for (int kk2 = 0; kk2 < 2; ++kk2) {
        v16bf bfr = load_frag_b(P_T, 64, kk2 * 32, mbase, lo, hi);
        for (int qt = 0; qt < 2; ++qt)
          a1[qt] = __builtin_amdgcn_wmma_f32_16x16x32_bf16(
              false, xfr[kk2][qt], false, bfr, (short)0, a1[qt], false, false);
      }
      // per lane: fixed m -> fixed (row2, c); 8 elements = consecutive q_local
      int m  = mbase + lo;                 // m = (o0,c,o1)
      int o0 = m >> 6, c = (m >> 3) & 7, o1 = m & 7;
      int row2 = o0 * 8 + o1;
      for (int qt = 0; qt < 2; ++qt) {
        uint4 pk = pk8_bf16(a1[qt]);
        *(uint4*)&T3A[row2 * 256 + c * 32 + qt * 16 + hi * 8] = pk;
      }
    }
    __syncthreads();

    // ---- Stage 2 partial: accumulate 256 of 512 reduction terms
    for (int kloc = 0; kloc < 256; kloc += 32) {
      v16bf bfr[4];
      for (int mt = 0; mt < 4; ++mt)
        bfr[mt] = load_frag_b(T3A, 256, kloc, mt * 16, lo, hi);
      for (int ni = 0; ni < 2; ++ni) {
        v16bf afr = load_frag_a(W2T, 512, (wave * 2 + ni) * 16,
                                rep * 256 + kloc, lo, hi);
        for (int mt = 0; mt < 4; ++mt)
          acc2[ni][mt] = __builtin_amdgcn_wmma_f32_16x16x32_bf16(
              false, afr, false, bfr[mt], (short)0, acc2[ni][mt], false, false);
      }
    }
    __syncthreads();   // before next rep overwrites T3A
  }

  // out[b, (o0,o1)=m2, (o2,o3,r)=n]; per lane: fixed m2, 8 consecutive n
  float* ob = out + (size_t)b * 32768;
  for (int ni = 0; ni < 2; ++ni) {
    int nbase = (wave * 2 + ni) * 16 + hi * 8;
    for (int mt = 0; mt < 4; ++mt) {
      int m2 = mt * 16 + lo;
      AccU u; u.v = acc2[ni][mt];
      *(float4*)(ob + m2 * 512 + nbase)     = u.f4[0];
      *(float4*)(ob + m2 * 512 + nbase + 4) = u.f4[1];
    }
  }
}

extern "C" void kernel_launch(void* const* d_in, const int* in_sizes, int n_in,
                              void* d_out, int out_size, void* d_ws, size_t ws_size,
                              hipStream_t stream) {
  const float* x       = (const float*)d_in[0];   // (512,8,8,8,8) f32
  const float* factors = (const float*)d_in[1];   // (4,8,8,8)     f32
  const float* cores   = (const float*)d_in[2];   // (3,8,8,8)     f32

  unsigned short* P_T = (unsigned short*)d_ws;                        // 64 KB
  unsigned short* W2T = (unsigned short*)((char*)d_ws + 65536);       // 512 KB

  ht_precompute<<<1, 512, 0, stream>>>(factors, cores, P_T, W2T);
  ht_forward<<<512, 512, 0, stream>>>(x, P_T, W2T, (float*)d_out);
}
